// FullyFusedDeformNetwork_2001454760056
// MI455X (gfx1250) — matrix operand order repaired and checked by reading, not compiled, not run on hardware
//
#include <hip/hip_runtime.h>

typedef _Float16 v8h  __attribute__((ext_vector_type(8)));
typedef _Float16 v16h __attribute__((ext_vector_type(16)));
typedef float    v8f  __attribute__((ext_vector_type(8)));
typedef unsigned int u32x4 __attribute__((ext_vector_type(4)));
typedef int          i32x4 __attribute__((ext_vector_type(4)));
typedef int          i32x8 __attribute__((ext_vector_type(8)));

#define THREADS 256
#define ROWS    128      // points per block
#define NHID    14       // hidden 128x128 layers
#define AS      136      // act LDS row stride (halves): 272B, bank-conflict-free
#define WS      136      // weight LDS row stride (halves)

// f16 swizzled weight arena in d_ws (halves), dense stride-128 rows [n][k]:
//   [0,            16384)        : W_in   (128 rows n, k 0..127, k>=88 zero)
//   [16384,        16384*15)     : W_hid  layer L at 16384 + L*16384
//   [16384*15,     16384*15+2048): W_out  (16 rows n (>=10 zero), k 0..127)
#define WQ_HID_OFF   16384
#define WQ_WOUT_OFF  (16384 * 15)
#define WQ_TOTAL     (16384 * 15 + 2048)

#if defined(__has_builtin)
#if __has_builtin(__builtin_amdgcn_tensor_load_to_lds)
#define HAVE_TDM 1
#endif
#endif

__global__ __launch_bounds__(THREADS) void swizzle_weights(
    const float* __restrict__ Win, const float* __restrict__ Whid,
    const float* __restrict__ Wout, _Float16* __restrict__ wq) {
  int i = blockIdx.x * THREADS + threadIdx.x;
  if (i >= WQ_TOTAL) return;
  const int k   = i & 127;
  const int row = i >> 7;
  float v;
  if (row < 128) {                       // W_in: global is (88,128) row-major (k,n)
    v = (k < 88) ? Win[k * 128 + row] : 0.0f;
  } else if (row < 128 * 15) {           // W_hid: (14,128,128) (L,k,n)
    const int L = (row - 128) >> 7;
    const int n = (row - 128) & 127;
    v = Whid[(L * 128 + k) * 128 + n];
  } else {                               // W_out: (128,10) (k,n)
    const int n = row - 128 * 15;
    v = (n < 10) ? Wout[k * 10 + n] : 0.0f;
  }
  wq[i] = (_Float16)v;
}

#if HAVE_TDM
// TDM: async DMA of a [rows x 256B] dense tile into LDS with a 16B pad after
// every 256B row -> LDS row stride 272B (= 136 halves), matching AS/WS.
__device__ __forceinline__ void tdm_load_tile(const _Float16* gsrc,
                                              unsigned lds_byte_off, int rows) {
  const unsigned long long ga = (unsigned long long)(uintptr_t)gsrc;
  u32x4 g0;
  g0[0] = 1u;                                             // count=1, user D#
  g0[1] = lds_byte_off;                                   // lds_addr
  g0[2] = (unsigned)(ga & 0xFFFFFFFFu);                   // global_addr[31:0]
  g0[3] = (unsigned)((ga >> 32) & 0x01FFFFFFu) | (2u << 30); // addr[56:32]|type=2
  i32x8 g1;
  g1[0] = (2 << 16)      // data_size = 4B
        | (1 << 20)      // pad_enable
        | (5 << 22)      // pad_interval: every 64 DWORDs (256B)
        | (3 << 25);     // pad_amount: 4 DWORDs (16B)
  g1[1] = (int)(64u << 16);            // tensor_dim0[15:0] = 64 DWORDs
  g1[2] = (int)((unsigned)rows << 16); // dim0 hi=0 | tensor_dim1[15:0] = rows
  g1[3] = (int)(64u << 16);            // dim1 hi=0 | tile_dim0 = 64 DWORDs
  g1[4] = rows;                        // tile_dim1 = rows, tile_dim2 = 0
  g1[5] = 64;                          // tensor_dim0_stride = 64 DWORDs
  g1[6] = (int)(8192u << 16);          // d0s hi=0 | tensor_dim1_stride lo
  g1[7] = 0;                           // tensor_dim1_stride hi
  const i32x4 z4 = {0, 0, 0, 0};
#if __clang_major__ >= 23
  const i32x8 z8 = {0, 0, 0, 0, 0, 0, 0, 0};
  __builtin_amdgcn_tensor_load_to_lds(g0, g1, z4, z4, z8, 0);
#else
  __builtin_amdgcn_tensor_load_to_lds(g0, g1, z4, z4, 0);
#endif
}
#endif

__global__ __launch_bounds__(THREADS) void fused_mlp(
    const float* __restrict__ x, const float* __restrict__ tt,
    const _Float16* __restrict__ wq, float* __restrict__ out, int Ntot) {
  __shared__ _Float16 act[ROWS * AS];
  __shared__ _Float16 wbuf[2][128 * WS];   // double-buffered weights

  const int tid  = threadIdx.x;
  const int lane = tid & 31;
  const int wave = tid >> 5;
  const int m    = lane & 15;   // row-in-tile (A/B) / column (C,D)
  const int hi   = lane >> 4;   // lane-half selector
  const int rowBase = blockIdx.x * ROWS;

#if HAVE_TDM
  // kick off W_in -> wbuf[0] and W_hid[0] -> wbuf[1] while we encode
  if (wave == 0) {
    tdm_load_tile(wq, (unsigned)(uintptr_t)&wbuf[0][0], 128);
    tdm_load_tile(wq + WQ_HID_OFF, (unsigned)(uintptr_t)&wbuf[1][0], 128);
  }
#else
  {
    const uint4* __restrict__ src = (const uint4*)wq;
    uint4* dst = (uint4*)&wbuf[0][0];
#pragma unroll
    for (int j = 0; j < 8; ++j) {
      const int i = j * THREADS + tid;
      dst[(i >> 4) * 17 + (i & 15)] = src[i];
    }
  }
#endif

  // ---- frequency encoding -> act[row][0..95] (cols 88..95 zero) ----
  {
    const int row = tid >> 1;
    const int q   = tid & 1;
    const int gr  = rowBase + row;
    const float x0 = x[gr * 3 + 0], x1 = x[gr * 3 + 1], x2 = x[gr * 3 + 2];
    const float tv = tt[gr];
    const float PI = 3.14159265358979323846f;
    _Float16* dst = &act[row * AS];
    for (int i = 0; i < 48; ++i) {
      const int c = q * 48 + i;
      float v;
      if (c < 3) {
        v = (c == 0) ? x0 : ((c == 1) ? x1 : x2);
      } else if (c < 75) {                 // sin block then cos block, (3,12)
        int idx = c - 3;
        const bool isCos = idx >= 36;
        if (isCos) idx -= 36;
        const int j = idx / 12, f = idx % 12;
        const float xv  = (j == 0) ? x0 : ((j == 1) ? x1 : x2);
        const float arg = xv * PI * (float)(1u << f);
        v = isCos ? __cosf(arg) : __sinf(arg);
      } else if (c == 75) {
        v = tv;
      } else if (c < 88) {
        int idx = c - 76;
        const bool isCos = idx >= 6;
        if (isCos) idx -= 6;
        const float arg = tv * PI * (float)(1u << idx);
        v = isCos ? __cosf(arg) : __sinf(arg);
      } else {
        v = 0.0f;
      }
      dst[c] = (_Float16)v;
    }
  }

#if HAVE_TDM
  if (wave == 0) __builtin_amdgcn_s_wait_tensorcnt(1);  // W_in landed
#endif
  __syncthreads();

  // ---- one dense layer: act(wave stripe) x wb -> relu -> act ----
  auto dense_relu = [&](const _Float16* wb, int kchunks) {
    v8f acc[8] = {};
    for (int kc = 0; kc < kchunks; ++kc) {
      // A fragment 16x32 f16: lane(m,hi) holds row m, K in {k0+8hi..+7, k0+16+8hi..+7}
      const _Float16* ap = &act[(wave * 16 + m) * AS + kc * 32 + hi * 8];
      const v8h a0 = *(const v8h*)(ap);
      const v8h a1 = *(const v8h*)(ap + 16);
      const v16h a = __builtin_shufflevector(a0, a1, 0,1,2,3,4,5,6,7,8,9,10,11,12,13,14,15);
#pragma unroll
      for (int nt = 0; nt < 8; ++nt) {
        // B fragment 32x16 f16: lane(m,hi) holds col n=nt*16+m, K in {k0+8hi..+7, k0+16+8hi..+7}
        const _Float16* bp = &wb[(nt * 16 + m) * WS + kc * 32 + hi * 8];
        const v8h b0 = *(const v8h*)(bp);
        const v8h b1 = *(const v8h*)(bp + 16);
        const v16h b = __builtin_shufflevector(b0, b1, 0,1,2,3,4,5,6,7,8,9,10,11,12,13,14,15);
        acc[nt] = __builtin_amdgcn_wmma_f32_16x16x32_f16(
            false, a, false, b, (short)0, acc[nt], false, false);
      }
    }
    // relu + f16 back into wave-private stripe (no cross-wave hazard)
#pragma unroll
    for (int nt = 0; nt < 8; ++nt)
#pragma unroll
      for (int r = 0; r < 8; ++r) {
        float v = acc[nt][r];
        v = v > 0.0f ? v : 0.0f;
        act[(wave * 16 + r + 8 * hi) * AS + nt * 16 + m] = (_Float16)v;
      }
  };

  dense_relu(&wbuf[0][0], 3);   // input layer: K = 96
  __syncthreads();              // everyone done reading wbuf[0]

  for (int L = 0; L < NHID; ++L) {
    const int cur = (L & 1) ^ 1;     // hid0 -> wbuf[1], hid1 -> wbuf[0], ...
    const int nxt = cur ^ 1;
#if HAVE_TDM
    if (wave == 0) {
      // buffer `nxt` was freed by the barrier above; start its DMA now
      if (L + 1 < NHID)
        tdm_load_tile(wq + WQ_HID_OFF + (L + 1) * 16384,
                      (unsigned)(uintptr_t)&wbuf[nxt][0], 128);
      else
        tdm_load_tile(wq + WQ_WOUT_OFF,
                      (unsigned)(uintptr_t)&wbuf[nxt][0], 16);
      __builtin_amdgcn_s_wait_tensorcnt(1);  // layer L weights have landed
    }
    __syncthreads();                         // publish wbuf[cur] to all waves
#else
    {
      const uint4* __restrict__ src = (const uint4*)(wq + WQ_HID_OFF + L * 16384);
      uint4* dst = (uint4*)&wbuf[cur][0];
#pragma unroll
      for (int j = 0; j < 8; ++j) {
        const int i = j * THREADS + tid;
        dst[(i >> 4) * 17 + (i & 15)] = src[i];
      }
    }
    __syncthreads();
#endif
    dense_relu(&wbuf[cur][0], 4);
    __syncthreads();
  }

  // ---- output layer: 128 -> 16 (10 valid); W_out sits in wbuf[1] ----
#if HAVE_TDM
  if (wave == 0) __builtin_amdgcn_s_wait_tensorcnt(0);
  __syncthreads();
#else
  {
    const uint4* __restrict__ src = (const uint4*)(wq + WQ_WOUT_OFF);
    uint4* dst = (uint4*)&wbuf[1][0];
    dst[(tid >> 4) * 17 + (tid & 15)] = src[tid];
  }
  __syncthreads();
#endif

  {
    v8f acc = {};
    for (int kc = 0; kc < 4; ++kc) {
      const _Float16* ap = &act[(wave * 16 + m) * AS + kc * 32 + hi * 8];
      const v8h a0 = *(const v8h*)(ap);
      const v8h a1 = *(const v8h*)(ap + 16);
      const v16h a = __builtin_shufflevector(a0, a1, 0,1,2,3,4,5,6,7,8,9,10,11,12,13,14,15);
      const _Float16* bp = &wbuf[1][m * WS + kc * 32 + hi * 8];
      const v8h b0 = *(const v8h*)(bp);
      const v8h b1 = *(const v8h*)(bp + 16);
      const v16h b = __builtin_shufflevector(b0, b1, 0,1,2,3,4,5,6,7,8,9,10,11,12,13,14,15);
      acc = __builtin_amdgcn_wmma_f32_16x16x32_f16(false, a, false, b, (short)0, acc, false, false);
    }
    // D layout: VGPR r, lane(m,hi): row = r + 8*hi, col = m
    const int col = m;
    const long long N3 = (long long)Ntot * 3;
    const long long N7 = (long long)Ntot * 7;
#pragma unroll
    for (int r = 0; r < 8; ++r) {
      const int grow = rowBase + wave * 16 + r + 8 * hi;
      const float v = acc[r];
      if (col < 3)       out[(long long)grow * 3 + col]            = v + 0.1f * x[grow * 3 + col];
      else if (col < 7)  out[N3 + (long long)grow * 4 + (col - 3)] = v;
      else if (col < 10) out[N7 + (long long)grow * 3 + (col - 7)] = v;
    }
  }
}

extern "C" void kernel_launch(void* const* d_in, const int* in_sizes, int n_in,
                              void* d_out, int out_size, void* d_ws, size_t ws_size,
                              hipStream_t stream) {
  (void)n_in; (void)out_size; (void)ws_size;
  const float* x    = (const float*)d_in[0];
  const float* t    = (const float*)d_in[1];
  const float* Win  = (const float*)d_in[2];
  const float* Whid = (const float*)d_in[3];
  const float* Wout = (const float*)d_in[4];
  float* out   = (float*)d_out;
  _Float16* wq = (_Float16*)d_ws;

  const int Ntot = in_sizes[0] / 3;   // 262144

  swizzle_weights<<<(WQ_TOTAL + THREADS - 1) / THREADS, THREADS, 0, stream>>>(
      Win, Whid, Wout, wq);
  fused_mlp<<<Ntot / ROWS, THREADS, 0, stream>>>(x, t, wq, out, Ntot);
}
